// MultiHeadAttention_33663953666247
// MI455X (gfx1250) — compile-verified, matrix operand order checked by reading, not run on hardware
//
#include <hip/hip_runtime.h>

// ---------------- constants (match reference) ----------------
#define B_  16
#define N_  512
#define D_  1024
#define H_  16
#define DK_ 64
#define PK_ 5
#define BN_ (B_ * N_)   // 8192 rows

typedef _Float16 h4  __attribute__((ext_vector_type(4)));
typedef _Float16 h8  __attribute__((ext_vector_type(8)));
typedef _Float16 v16h __attribute__((ext_vector_type(16)));
typedef float    v8f  __attribute__((ext_vector_type(8)));

static __device__ __forceinline__ v16h cat8(h8 lo, h8 hi) {
  v16h r;
#pragma unroll
  for (int i = 0; i < 8; ++i) { r[i] = lo[i]; r[i + 8] = hi[i]; }
  return r;
}

static __device__ __forceinline__ v8f wmma16(v16h a, v16h b, v8f c) {
  // D = A(16x32 f16) * B(32x16 f16) + C(16x16 f32)
  return __builtin_amdgcn_wmma_f32_16x16x32_f16(
      false, a, false, b, (short)0, c, false, false);
}

// CDNA5 async copy: global -> LDS, 16 bytes per lane, tracked by ASYNCcnt.
// GVS mode: mem_addr = SGPR base + VGPR 32-bit byte offset.
static __device__ __forceinline__ void async_cp_b128(
    _Float16* lds_dst, const _Float16* gbase, unsigned byte_off) {
  const unsigned lds_off = (unsigned)(uintptr_t)lds_dst;  // low 32b = LDS addr
  asm volatile("global_load_async_to_lds_b128 %0, %1, %2"
               :
               : "v"(lds_off), "v"(byte_off), "s"(gbase)
               : "memory");
}
static __device__ __forceinline__ void wait_async0() {
  asm volatile("s_wait_asynccnt 0" ::: "memory");
}

// ---------------- LayerNorm: x -> xn(f32) + xn(f16) ----------------
__global__ __launch_bounds__(256) void ln_kernel(
    const float* __restrict__ x, const float* __restrict__ g,
    const float* __restrict__ bta, float* __restrict__ xn,
    _Float16* __restrict__ xn16) {
  const int row = blockIdx.x;
  const int t = threadIdx.x;
  __shared__ float red[256];
  const float* xr = x + (size_t)row * D_;
  float a0 = xr[t], a1 = xr[t + 256], a2 = xr[t + 512], a3 = xr[t + 768];
  red[t] = a0 + a1 + a2 + a3;
  __syncthreads();
  for (int o = 128; o > 0; o >>= 1) {
    if (t < o) red[t] += red[t + o];
    __syncthreads();
  }
  const float mu = red[0] * (1.0f / (float)D_);
  __syncthreads();
  float d0 = a0 - mu, d1 = a1 - mu, d2 = a2 - mu, d3 = a3 - mu;
  red[t] = d0 * d0 + d1 * d1 + d2 * d2 + d3 * d3;
  __syncthreads();
  for (int o = 128; o > 0; o >>= 1) {
    if (t < o) red[t] += red[t + o];
    __syncthreads();
  }
  const float inv = rsqrtf(red[0] * (1.0f / (float)D_) + 1e-6f);
  float dv[4] = {d0, d1, d2, d3};
#pragma unroll
  for (int k = 0; k < 4; ++k) {
    const int c = t + 256 * k;
    const float y = dv[k] * inv * g[c] + bta[c];
    xn[(size_t)row * D_ + c] = y;
    xn16[(size_t)row * D_ + c] = (_Float16)y;
  }
}

// ---------------- weight f32 -> f16 with transpose: dst[n*K+k]=src[k*N+n] ---
__global__ __launch_bounds__(256) void convT_kernel(
    const float* __restrict__ src, _Float16* __restrict__ dst, int rows,
    int cols) {
  const int idx = blockIdx.x * 256 + threadIdx.x;
  if (idx < rows * cols) {
    const int k = idx / cols, n = idx % cols;
    dst[(size_t)n * rows + k] = (_Float16)src[idx];
  }
}

// ---------------- tiled WMMA GEMM with async double-buffered staging --------
// C(MxNout) = act(A(MxK) @ W(KxNout) + bias) [+resid]
// WT is the K-transposed weight: WT[n*K + k] = W[k*Nout + n]
// block tile 128x128, 256 threads (8 waves, each 4x2 = 8 wmma tiles)
template <int ACT, int HAS_BIAS, int OUT16, int RESID>
__global__ __launch_bounds__(256) void gemm128_kernel(
    const _Float16* __restrict__ A, const _Float16* __restrict__ WT,
    const float* __restrict__ bias, const float* __restrict__ resid,
    _Float16* __restrict__ C16, float* __restrict__ C32, int M, int K,
    int Nout) {
  __shared__ __align__(16) _Float16 As[2][128][32];
  __shared__ __align__(16) _Float16 Ws[2][128][32];
  const int t = threadIdx.x, wave = t >> 5, lane = t & 31;
  const int half = lane >> 4, nr = lane & 15;
  const int m0 = blockIdx.y * 128, n0 = blockIdx.x * 128;
  const int wm = wave >> 2, wn = wave & 3;  // wave tile: 64 rows x 32 cols

  // staging map: 512 16B-chunks per tile, 2 per thread
  const int r0 = t >> 2, c0 = (t & 3) * 8;          // chunk t
  const int r1 = (t + 256) >> 2, c1 = ((t + 256) & 3) * 8;  // chunk t+256
  const _Float16* Ab = A + (size_t)m0 * K;
  const _Float16* Wb = WT + (size_t)n0 * K;
  const unsigned a_off0 = (unsigned)((r0 * K + c0) * 2);
  const unsigned a_off1 = (unsigned)((r1 * K + c1) * 2);

  v8f acc[4][2];
#pragma unroll
  for (int i = 0; i < 4; ++i)
#pragma unroll
    for (int j = 0; j < 2; ++j)
      acc[i][j] = (v8f){0.f, 0.f, 0.f, 0.f, 0.f, 0.f, 0.f, 0.f};

  const int NSTEP = K / 32;
  // prefetch step 0 into buffer 0
  async_cp_b128(&As[0][r0][c0], Ab, a_off0);
  async_cp_b128(&As[0][r1][c1], Ab, a_off1);
  async_cp_b128(&Ws[0][r0][c0], Wb, a_off0);
  async_cp_b128(&Ws[0][r1][c1], Wb, a_off1);

  for (int ks = 0; ks < NSTEP; ++ks) {
    const int cur = ks & 1, nxt = cur ^ 1;
    wait_async0();     // this wave's copies into buf[cur] complete
    __syncthreads();   // all waves' copies done; all done reading buf[nxt]
    if (ks + 1 < NSTEP) {
      const unsigned kb = (unsigned)((ks + 1) * 64);  // 32 halves = 64 bytes
      async_cp_b128(&As[nxt][r0][c0], Ab, a_off0 + kb);
      async_cp_b128(&As[nxt][r1][c1], Ab, a_off1 + kb);
      async_cp_b128(&Ws[nxt][r0][c0], Wb, a_off0 + kb);
      async_cp_b128(&Ws[nxt][r1][c1], Wb, a_off1 + kb);
    }
    // fragments
    v16h af[4], bf[2];
#pragma unroll
    for (int rt = 0; rt < 4; ++rt) {
      const int row = wm * 64 + rt * 16 + nr;
      af[rt] = cat8(*(const h8*)&As[cur][row][half * 8],
                    *(const h8*)&As[cur][row][16 + half * 8]);
    }
#pragma unroll
    for (int ct = 0; ct < 2; ++ct) {
      const int col = wn * 32 + ct * 16 + nr;
      bf[ct] = cat8(*(const h8*)&Ws[cur][col][half * 8],
                    *(const h8*)&Ws[cur][col][16 + half * 8]);
    }
#pragma unroll
    for (int rt = 0; rt < 4; ++rt)
#pragma unroll
      for (int ct = 0; ct < 2; ++ct)
        acc[rt][ct] = wmma16(af[rt], bf[ct], acc[rt][ct]);
  }

  // epilogue
#pragma unroll
  for (int rt = 0; rt < 4; ++rt) {
#pragma unroll
    for (int ct = 0; ct < 2; ++ct) {
      const int col = n0 + wn * 32 + ct * 16 + nr;
      const float bv = HAS_BIAS ? bias[col] : 0.0f;
#pragma unroll
      for (int r = 0; r < 8; ++r) {
        const int row = m0 + wm * 64 + rt * 16 + r + 8 * half;
        float v = acc[rt][ct][r] + bv;
        if (ACT == 1) v = v > 0.0f ? v : 0.0f;
        if (ACT == 2) v = 1.0f / (1.0f + __expf(-v));
        const size_t o = (size_t)row * Nout + col;
        if (RESID) v += resid[o];
        if (OUT16) C16[o] = (_Float16)v;
        else C32[o] = v;
      }
    }
  }
}

// ---------------- rel[b,h,i,p] = sum_d q[b,i,h,d] * rel_k[p,d] ----------------
__global__ __launch_bounds__(256) void rel_kernel(
    const _Float16* __restrict__ q16, const float* __restrict__ rel_k,
    float* __restrict__ rel) {
  const int idx = blockIdx.x * 256 + threadIdx.x;
  if (idx >= B_ * H_ * N_ * PK_) return;
  const int p = idx % PK_;
  const int i = (idx / PK_) % N_;
  const int h = (idx / (PK_ * N_)) % H_;
  const int b = idx / (PK_ * N_ * H_);
  const _Float16* qp = q16 + (((size_t)b * N_ + i) * H_ + h) * DK_;
  const float* rp = rel_k + (size_t)p * DK_;
  float s = 0.0f;
#pragma unroll 8
  for (int d = 0; d < DK_; ++d) s += (float)qp[d] * rp[d];
  rel[(((size_t)b * H_ + h) * N_ + i) * PK_ + p] = s;
}

// ---------------- attention: per (b, h, 16-row strip) ----------------
__global__ __launch_bounds__(256) void attn_kernel(
    const _Float16* __restrict__ q16, const _Float16* __restrict__ k16,
    const _Float16* __restrict__ v16, const float* __restrict__ rel,
    const int* __restrict__ gpm, const unsigned char* __restrict__ mask,
    float* __restrict__ attn_out, _Float16* __restrict__ ao16) {
  __shared__ __align__(16) _Float16 Qs[16][64];
  __shared__ __align__(16) float Ssh[16][512];
  __shared__ __align__(16) _Float16 Ps[16][512];
  __shared__ float red[16][17];
  const int t = threadIdx.x, wave = t >> 5, lane = t & 31;
  const int half = lane >> 4, nr = lane & 15;
  const int it = blockIdx.x, h = blockIdx.y, b = blockIdx.z;
  const int i0 = it * 16;

  {
    const int r = t >> 4, c = (t & 15) * 4;
    *(h4*)&Qs[r][c] =
        *(const h4*)&q16[(((size_t)b * N_ + i0 + r) * H_ + h) * DK_ + c];
  }
  __syncthreads();

  // S = Q K^T : 32 column tiles, 4 per wave
  for (int jt = wave; jt < 32; jt += 8) {
    v8f acc = {0.f, 0.f, 0.f, 0.f, 0.f, 0.f, 0.f, 0.f};
    const int j = jt * 16 + nr;
    const _Float16* kp = k16 + (((size_t)b * N_ + j) * H_ + h) * DK_;
#pragma unroll
    for (int s = 0; s < 2; ++s) {
      const v16h a = cat8(*(const h8*)&Qs[nr][s * 32 + half * 8],
                          *(const h8*)&Qs[nr][s * 32 + 16 + half * 8]);
      const v16h bb = cat8(*(const h8*)&kp[s * 32 + half * 8],
                           *(const h8*)&kp[s * 32 + 16 + half * 8]);
      acc = wmma16(a, bb, acc);
    }
#pragma unroll
    for (int r = 0; r < 8; ++r) Ssh[r + 8 * half][jt * 16 + nr] = acc[r];
  }
  __syncthreads();

  // rel bias gather + scale + mask
  const float scale = 0.125f;  // 1/sqrt(64)
#pragma unroll
  for (int e = 0; e < 32; ++e) {
    const int idx = t + 256 * e;
    const int i = idx >> 9, j = idx & 511;
    float v = Ssh[i][j];
    const int p = gpm[((size_t)b * N_ + i0 + i) * N_ + j];
    v = (v + rel[(((size_t)b * H_ + h) * N_ + (i0 + i)) * PK_ + p]) * scale;
    if (!mask[b * N_ + j]) v = -9e9f;
    Ssh[i][j] = v;
  }
  __syncthreads();

  // softmax (16 threads per row)
  const int row = t >> 4, sub = t & 15;
  {
    float m = -3.4e38f;
    for (int q = 0; q < 32; ++q) m = fmaxf(m, Ssh[row][sub + 16 * q]);
    red[row][sub] = m;
  }
  __syncthreads();
  float rmax = red[row][0];
#pragma unroll
  for (int i = 1; i < 16; ++i) rmax = fmaxf(rmax, red[row][i]);
  __syncthreads();
  {
    float s = 0.0f;
    for (int q = 0; q < 32; ++q) {
      const int j = sub + 16 * q;
      const float e = __expf(Ssh[row][j] - rmax);
      Ssh[row][j] = e;
      s += e;
    }
    red[row][sub] = s;
  }
  __syncthreads();
  {
    float s = 0.0f;
#pragma unroll
    for (int i = 0; i < 16; ++i) s += red[row][i];
    const float inv = 1.0f / s;
    for (int q = 0; q < 32; ++q) {
      const int j = sub + 16 * q;
      const float p = Ssh[row][j] * inv;
      attn_out[(((size_t)b * H_ + h) * N_ + i0 + row) * N_ + j] = p;
      Ps[row][j] = (_Float16)p;
    }
  }
  __syncthreads();

  // O = P @ V : 4 d-tiles of 16, waves 0..3 (wave-uniform, EXEC all ones)
  if (wave < 4) {
    const int dt = wave;
    v8f acc = {0.f, 0.f, 0.f, 0.f, 0.f, 0.f, 0.f, 0.f};
    for (int ks = 0; ks < 16; ++ks) {
      const int jb = ks * 32;
      const v16h a = cat8(*(const h8*)&Ps[nr][jb + half * 8],
                          *(const h8*)&Ps[nr][jb + 16 + half * 8]);
      v16h bb;
#pragma unroll
      for (int u = 0; u < 16; ++u) {
        const int kk = (u < 8) ? (half * 8 + u) : (16 + half * 8 + (u - 8));
        bb[u] = v16[(((size_t)b * N_ + jb + kk) * H_ + h) * DK_ + dt * 16 + nr];
      }
      acc = wmma16(a, bb, acc);
    }
#pragma unroll
    for (int r = 0; r < 8; ++r) {
      const int orow = i0 + r + 8 * half;
      ao16[(((size_t)b * N_ + orow) * H_ + h) * DK_ + dt * 16 + nr] =
          (_Float16)acc[r];
    }
  }
}

// ---------------- gate multiply: ag16 = ao16 * sigmoid-gate(f32) ----------------
__global__ __launch_bounds__(256) void mulgate_kernel(
    const _Float16* __restrict__ ao16, const float* __restrict__ gate,
    _Float16* __restrict__ ag16, int n) {
  const int i = blockIdx.x * 256 + threadIdx.x;
  if (i < n) ag16[i] = (_Float16)((float)ao16[i] * gate[i]);
}

// ---------------- launch ----------------
extern "C" void kernel_launch(void* const* d_in, const int* in_sizes, int n_in,
                              void* d_out, int out_size, void* d_ws,
                              size_t ws_size, hipStream_t stream) {
  const float* x     = (const float*)d_in[0];
  const int*   gpm   = (const int*)d_in[1];
  const unsigned char* mask = (const unsigned char*)d_in[2];
  const float* w_q = (const float*)d_in[3];
  const float* w_k = (const float*)d_in[4];
  const float* w_v1 = (const float*)d_in[5];
  const float* b_v1 = (const float*)d_in[6];
  const float* w_v2 = (const float*)d_in[7];
  const float* b_v2 = (const float*)d_in[8];
  const float* w_g = (const float*)d_in[9];
  const float* b_g = (const float*)d_in[10];
  const float* w_o = (const float*)d_in[11];
  const float* b_o = (const float*)d_in[12];
  const float* ln_g = (const float*)d_in[13];
  const float* ln_b = (const float*)d_in[14];
  const float* rel_k = (const float*)d_in[15];

  float* out_f32 = (float*)d_out;                   // (B,N,D) f32
  float* attn_out = out_f32 + (size_t)BN_ * D_;     // (B,H,N,N) f32

  // workspace carve-out
  char* p = (char*)d_ws;
  auto carve = [&](size_t bytes) {
    void* r = (void*)p;
    p += (bytes + 255) & ~(size_t)255;
    return r;
  };
  float*    xn    = (float*)carve((size_t)BN_ * D_ * 4);
  _Float16* xn16  = (_Float16*)carve((size_t)BN_ * D_ * 2);
  _Float16* q16   = (_Float16*)carve((size_t)BN_ * D_ * 2);
  _Float16* k16   = (_Float16*)carve((size_t)BN_ * D_ * 2);
  _Float16* h16   = (_Float16*)carve((size_t)BN_ * D_ * 2);
  _Float16* v16   = (_Float16*)carve((size_t)BN_ * D_ * 2);
  float*    gate  = (float*)carve((size_t)BN_ * D_ * 4);
  _Float16* ao16  = (_Float16*)carve((size_t)BN_ * D_ * 2);
  float*    rel   = (float*)carve((size_t)B_ * H_ * N_ * PK_ * 4);
  _Float16* wq16  = (_Float16*)carve((size_t)D_ * D_ * 2);   // transposed
  _Float16* wk16  = (_Float16*)carve((size_t)D_ * D_ * 2);
  _Float16* wv116 = (_Float16*)carve((size_t)D_ * D_ * 2);
  _Float16* wv216 = (_Float16*)carve((size_t)D_ * D_ * 2);
  _Float16* wg16  = (_Float16*)carve((size_t)D_ * D_ * 2);
  _Float16* wo16  = (_Float16*)carve((size_t)D_ * D_ * 2);
  _Float16* ag16  = xn16;  // reuse: xn16 dead after gate GEMM

  // 1) LayerNorm
  ln_kernel<<<BN_, 256, 0, stream>>>(x, ln_g, ln_b, xn, xn16);

  // 2) convert + transpose weights to f16 (WT[n][k])
  const int wn = D_ * D_, wblk = (wn + 255) / 256;
  convT_kernel<<<wblk, 256, 0, stream>>>(w_q, wq16, D_, D_);
  convT_kernel<<<wblk, 256, 0, stream>>>(w_k, wk16, D_, D_);
  convT_kernel<<<wblk, 256, 0, stream>>>(w_v1, wv116, D_, D_);
  convT_kernel<<<wblk, 256, 0, stream>>>(w_v2, wv216, D_, D_);
  convT_kernel<<<wblk, 256, 0, stream>>>(w_g, wg16, D_, D_);
  convT_kernel<<<wblk, 256, 0, stream>>>(w_o, wo16, D_, D_);

  // 3) projection GEMMs (128x128 tiles, async double-buffered)
  dim3 gg(D_ / 128, BN_ / 128);
  gemm128_kernel<0, 0, 1, 0><<<gg, 256, 0, stream>>>(
      xn16, wq16, nullptr, nullptr, q16, nullptr, BN_, D_, D_);
  gemm128_kernel<0, 0, 1, 0><<<gg, 256, 0, stream>>>(
      xn16, wk16, nullptr, nullptr, k16, nullptr, BN_, D_, D_);
  gemm128_kernel<1, 1, 1, 0><<<gg, 256, 0, stream>>>(   // relu(xn@w_v1+b_v1)
      xn16, wv116, b_v1, nullptr, h16, nullptr, BN_, D_, D_);
  gemm128_kernel<0, 1, 1, 0><<<gg, 256, 0, stream>>>(   // h@w_v2+b_v2
      h16, wv216, b_v2, nullptr, v16, nullptr, BN_, D_, D_);
  gemm128_kernel<2, 1, 0, 0><<<gg, 256, 0, stream>>>(   // sigmoid(xn@w_g+b_g)
      xn16, wg16, b_g, nullptr, nullptr, gate, BN_, D_, D_);

  // 4) relative-position projection
  const int rn = B_ * H_ * N_ * PK_;
  rel_kernel<<<(rn + 255) / 256, 256, 0, stream>>>(q16, rel_k, rel);

  // 5) attention
  attn_kernel<<<dim3(N_ / 16, H_, B_), 256, 0, stream>>>(
      q16, k16, v16, rel, gpm, mask, attn_out, ao16);

  // 6) gate multiply
  const int gn = BN_ * D_;
  mulgate_kernel<<<(gn + 255) / 256, 256, 0, stream>>>(ao16, gate, ag16, gn);

  // 7) final GEMM with residual: out = xn + (ag @ w_o + b_o)
  gemm128_kernel<0, 1, 0, 1><<<gg, 256, 0, stream>>>(
      ag16, wo16, b_o, xn, nullptr, out_f32, BN_, D_, D_);

  (void)in_sizes; (void)n_in; (void)out_size; (void)ws_size;
}
